// SMoE_47476568490359
// MI455X (gfx1250) — compile-verified, matrix operand order checked by reading, not run on hardware
//
#include <hip/hip_runtime.h>

// ---------------- problem constants ----------------
constexpr int N = 4096;
constexpr int D = 1024;
constexpr int O = 1024;
constexpr int E = 16;
constexpr float EPS  = 0.2f;
constexpr float GEPS = 0.2f;

// ---------------- CDNA5 WMMA types ----------------
typedef __bf16 bf16_t;
typedef __attribute__((ext_vector_type(8)))  __bf16 bf16x8;
typedef __attribute__((ext_vector_type(16))) __bf16 v16bf;
typedef __attribute__((ext_vector_type(8)))  float  v8f;
typedef __attribute__((ext_vector_type(4)))  int    v4i;

union AFrag { v16bf v; bf16x8 h[2]; };

// ---------------- CDNA5 async global->LDS helpers ----------------
#if defined(__has_builtin)
#if __has_builtin(__builtin_amdgcn_global_load_async_to_lds_b128)
#define HAVE_ASYNC_LDS 1
#endif
#if __has_builtin(__builtin_amdgcn_s_wait_asynccnt)
#define HAVE_WAIT_ASYNC 1
#endif
#endif

__device__ __forceinline__ void asyncCopy16(const bf16_t* g, bf16_t* l) {
#ifdef HAVE_ASYNC_LDS
    __builtin_amdgcn_global_load_async_to_lds_b128(
        (__attribute__((address_space(1))) v4i*)g,
        (__attribute__((address_space(3))) v4i*)l, 0, 0);
#else
    *(bf16x8*)l = *(const bf16x8*)g;
#endif
}

__device__ __forceinline__ void waitAsync() {
#ifdef HAVE_ASYNC_LDS
#ifdef HAVE_WAIT_ASYNC
    __builtin_amdgcn_s_wait_asynccnt(0);
#else
    asm volatile("s_wait_asynccnt 0x0" ::: "memory");
#endif
#endif
}

// =====================================================================
// Kernel 1: selector logits + softmax + per-token sparse gating.
// One wave32 per token; butterfly reduce; E=16 serial gating math.
// =====================================================================
__global__ __launch_bounds__(256) void selector_kernel(
    const float* __restrict__ x, const float* __restrict__ Wsel,
    const float* __restrict__ bsel, float* __restrict__ w,
    float* __restrict__ sw2, unsigned* __restrict__ u1bits,
    float* __restrict__ softCost)
{
    const int wave = threadIdx.x >> 5;
    const int lane = threadIdx.x & 31;
    const int n = blockIdx.x * 8 + wave;
    if (n >= N) return;

    const float* xrow = x + (size_t)n * D;
    float xv[32];
#pragma unroll
    for (int j = 0; j < 32; ++j) xv[j] = xrow[j * 32 + lane];

    float logit[E];
#pragma unroll
    for (int e = 0; e < E; ++e) {
        const float* wr = Wsel + (size_t)e * D;
        float acc = 0.0f;
#pragma unroll
        for (int j = 0; j < 32; ++j) acc = fmaf(xv[j], wr[j * 32 + lane], acc);
#pragma unroll
        for (int off = 16; off > 0; off >>= 1) acc += __shfl_xor(acc, off, 32);
        logit[e] = acc + bsel[e];
    }

    float mx = logit[0];
#pragma unroll
    for (int e = 1; e < E; ++e) mx = fmaxf(mx, logit[e]);
    float p[E], s = 0.0f;
#pragma unroll
    for (int e = 0; e < E; ++e) { p[e] = __expf(logit[e] - mx); s += p[e]; }
    float wv[E];
    const float inv = 1.0f / s;
#pragma unroll
    for (int e = 0; e < E; ++e) wv[e] = p[e] * inv;

    // descending insertion sort with argsort indices
    float sv[E]; int ord[E];
#pragma unroll
    for (int e = 0; e < E; ++e) { sv[e] = wv[e]; ord[e] = e; }
    for (int i = 1; i < E; ++i) {
        float kv = sv[i]; int ki = ord[i]; int j = i - 1;
        while (j >= 0 && sv[j] < kv) { sv[j + 1] = sv[j]; ord[j + 1] = ord[j]; --j; }
        sv[j + 1] = kv; ord[j + 1] = ki;
    }

    float cum[E];
    cum[0] = sv[0];
    for (int e = 1; e < E; ++e) cum[e] = cum[e - 1] + sv[e];

    float sw[E];
#pragma unroll
    for (int e = 0; e < E; ++e) {
        float shifted = (e < E - 1) ? cum[e + 1] : 1.0f;
        sw[e] = fmaxf(0.0f, fminf(shifted, 1.0f - EPS) - cum[e]);
    }
    float sc = 0.0f;
#pragma unroll
    for (int e = 0; e < E; ++e) {
        float nxt = (e < E - 1) ? sw[e + 1] : 0.0f;
        sc += (nxt > 0.0f) ? 1.0f : sw[e];
    }
    float sw2v[E];
    unsigned bits = 0;
    for (int e = 0; e < E; ++e) {
        sw2v[e] = sw[ord[e]];               // faithful gather quirk
        if (sw2v[e] > 0.0f) bits |= (1u << e);
    }

    if (lane == 0) {
        float* wr = w + (size_t)n * E;
        float* s2 = sw2 + (size_t)n * E;
#pragma unroll
        for (int e = 0; e < E; ++e) { wr[e] = wv[e]; s2[e] = sw2v[e]; }
        u1bits[n] = bits;
        softCost[n] = sc;
    }
}

// =====================================================================
// Kernel 2: deterministic per-expert column sum of w.
// =====================================================================
__global__ __launch_bounds__(256) void colsum_kernel(
    const float* __restrict__ w, float* __restrict__ colsum)
{
    __shared__ float red[256];
    const int e = blockIdx.x;
    float s = 0.0f;
    for (int n = threadIdx.x; n < N; n += 256) s += w[(size_t)n * E + e];
    red[threadIdx.x] = s;
    __syncthreads();
    for (int o = 128; o > 0; o >>= 1) {
        if (threadIdx.x < o) red[threadIdx.x] += red[threadIdx.x + o];
        __syncthreads();
    }
    if (threadIdx.x == 0) colsum[e] = red[0];
}

// =====================================================================
// Kernel 3: per-expert token sort (bitonic in LDS), scan, threshold,
// gather-quirk usage bits. One 1024-thread workgroup per expert.
// =====================================================================
__global__ __launch_bounds__(1024) void expert_sort_kernel(
    const float* __restrict__ w, const float* __restrict__ colsum,
    unsigned* __restrict__ u2bits)
{
    __shared__ float key[N];
    __shared__ int   idx[N];
    __shared__ float tmp[N];
    const int e = blockIdx.x;
    const float inv = 1.0f / colsum[e];

    for (int t = threadIdx.x; t < N; t += 1024) {
        key[t] = w[(size_t)t * E + e] * inv;
        idx[t] = t;
    }
    __syncthreads();

    for (int k = 2; k <= N; k <<= 1) {
        for (int j = k >> 1; j > 0; j >>= 1) {
            for (int t = threadIdx.x; t < N; t += 1024) {
                int p = t ^ j;
                if (p > t) {
                    bool desc = ((t & k) == 0);
                    float a = key[t], b = key[p];
                    if ((a < b) == desc) {
                        key[t] = b; key[p] = a;
                        int ia = idx[t]; idx[t] = idx[p]; idx[p] = ia;
                    }
                }
            }
            __syncthreads();
        }
    }

    for (int off = 1; off < N; off <<= 1) {
        for (int t = threadIdx.x; t < N; t += 1024)
            tmp[t] = key[t] + ((t >= off) ? key[t - off] : 0.0f);
        __syncthreads();
        for (int t = threadIdx.x; t < N; t += 1024) key[t] = tmp[t];
        __syncthreads();
    }

    for (int t = threadIdx.x; t < N; t += 1024) {
        float shifted = (t < N - 1) ? key[t + 1] : 1.0f;
        tmp[t] = ((fminf(shifted, 1.0f - GEPS) - key[t]) > 0.0f) ? 1.0f : 0.0f;
    }
    __syncthreads();

    for (int t = threadIdx.x; t < N; t += 1024) {
        int r = idx[t];                      // faithful gather quirk
        if (tmp[r] != 0.0f) atomicOr(&u2bits[t], 1u << e);
    }
}

// =====================================================================
// Kernel 4: combine usage masks into final routing weights.
// =====================================================================
__global__ __launch_bounds__(256) void finalize_su(
    const float* __restrict__ sw2, const unsigned* __restrict__ u1,
    const unsigned* __restrict__ u2, float* __restrict__ su)
{
    int i = blockIdx.x * 256 + threadIdx.x;
    if (i >= N * E) return;
    int n = i >> 4, e = i & 15;
    unsigned bits = u1[n] | u2[n];
    su[i] = ((bits >> e) & 1u) ? sw2[i] : 0.0f;
}

// =====================================================================
// Kernel 5: f32 -> bf16 conversion (8 elements/thread, packed).
// =====================================================================
__global__ __launch_bounds__(256) void cvt_bf16_kernel(
    const float* __restrict__ src, bf16_t* __restrict__ dst, int n8)
{
    int i = blockIdx.x * 256 + threadIdx.x;
    if (i >= n8) return;
    const float4* s = (const float4*)src + (size_t)i * 2;
    float4 a = s[0], b = s[1];
    bf16x8 o;
    o[0] = (bf16_t)a.x; o[1] = (bf16_t)a.y; o[2] = (bf16_t)a.z; o[3] = (bf16_t)a.w;
    o[4] = (bf16_t)b.x; o[5] = (bf16_t)b.y; o[6] = (bf16_t)b.z; o[7] = (bf16_t)b.w;
    *((bf16x8*)dst + i) = o;
}

// =====================================================================
// Kernel 6: sparse grouped GEMM via bf16 WMMA with double-buffered
// async global->LDS staging. Block: 128x128 tile, 8 wave32s (2x4).
// =====================================================================
constexpr int BM = 128, BN = 128, BK = 32;
constexpr int LDA = 40;  // padded bf16 row stride (80B, 16B aligned)

__device__ __forceinline__ void issueTile(const bf16_t* __restrict__ g,
                                          bf16_t* __restrict__ l,
                                          int kt, int tid)
{
    // 128 rows x 32 bf16 = 512 chunks of 16B; 256 threads -> 2 chunks each
#pragma unroll
    for (int v = 0; v < 2; ++v) {
        int c   = tid + v * 256;
        int row = c >> 2;
        int off = (c & 3) * 8;
        asyncCopy16(g + (size_t)row * D + kt + off, l + row * LDA + off);
    }
}

__global__ __launch_bounds__(256, 1) void moe_gemm(
    const bf16_t* __restrict__ xb, const bf16_t* __restrict__ Wb,
    const float* __restrict__ bexp, const float* __restrict__ su,
    float* __restrict__ out)
{
    __shared__ __align__(16) bf16_t At[2][BM * LDA];
    __shared__ __align__(16) bf16_t Bt[2][BN * LDA];
    __shared__ float suLds[BM * E];
    __shared__ float bexpLds[BN];
    __shared__ unsigned activeMask;

    const int tid   = threadIdx.x;
    const int mBase = blockIdx.y * BM;
    const int oBase = blockIdx.x * BN;

    if (tid == 0) activeMask = 0;
    __syncthreads();
    unsigned localBits = 0;
    for (int i = tid; i < BM * E; i += 256) {
        float v = su[(size_t)(mBase + (i >> 4)) * E + (i & 15)];
        suLds[i] = v;
        if (v != 0.0f) localBits |= (1u << (i & 15));
    }
    if (localBits) atomicOr(&activeMask, localBits);
    __syncthreads();
    const unsigned amask = activeMask;

    const int wave  = tid >> 5;
    const int lane  = tid & 31;
    const int mw    = (wave >> 2) * 64;   // 0 or 64
    const int nw    = (wave & 3) * 32;    // 0,32,64,96
    const int laneM = lane & 15;
    const int kGrp  = (lane >> 4) * 8;    // 0 or 8 (bf16 elements)
    const int mHalf = (lane >> 4) * 8;    // C-layout row offset for hi lanes

    const bf16_t* xg = xb + (size_t)mBase * D;

    v8f outAcc[4][2];
#pragma unroll
    for (int a = 0; a < 4; ++a)
#pragma unroll
        for (int b = 0; b < 2; ++b)
#pragma unroll
            for (int i = 0; i < 8; ++i) outAcc[a][b][i] = 0.0f;

    for (int e = 0; e < E; ++e) {
        if (!((amask >> e) & 1u)) continue;
        __syncthreads();   // previous expert's LDS reads done
        for (int i = tid; i < BN; i += 256)
            bexpLds[i] = bexp[(size_t)e * O + oBase + i];
        const bf16_t* wg = Wb + ((size_t)e * O + oBase) * D;

        v8f cAcc[4][2];
#pragma unroll
        for (int a = 0; a < 4; ++a)
#pragma unroll
            for (int b = 0; b < 2; ++b)
#pragma unroll
                for (int i = 0; i < 8; ++i) cAcc[a][b][i] = 0.0f;

        // prologue: stage K-slice 0 into buffer 0
        issueTile(xg, At[0], 0, tid);
        issueTile(wg, Bt[0], 0, tid);
        waitAsync();
        __syncthreads();

        for (int kt = 0; kt < D; kt += BK) {
            const int cur = (kt / BK) & 1;
            if (kt + BK < D) {               // stream next slice into other buf
                issueTile(xg, At[cur ^ 1], kt + BK, tid);
                issueTile(wg, Bt[cur ^ 1], kt + BK, tid);
            }
            const bf16_t* Ac = At[cur];
            const bf16_t* Bc = Bt[cur];

            AFrag bF[2];
#pragma unroll
            for (int nsub = 0; nsub < 2; ++nsub) {
                int o = nw + nsub * 16 + laneM;
                bF[nsub].h[0] = *(const bf16x8*)&Bc[o * LDA + kGrp];
                bF[nsub].h[1] = *(const bf16x8*)&Bc[o * LDA + kGrp + 16];
            }
#pragma unroll
            for (int msub = 0; msub < 4; ++msub) {
                int r = mw + msub * 16 + laneM;
                AFrag aF;
                aF.h[0] = *(const bf16x8*)&Ac[r * LDA + kGrp];
                aF.h[1] = *(const bf16x8*)&Ac[r * LDA + kGrp + 16];
#pragma unroll
                for (int nsub = 0; nsub < 2; ++nsub) {
                    cAcc[msub][nsub] = __builtin_amdgcn_wmma_f32_16x16x32_bf16(
                        false, aF.v, false, bF[nsub].v, (short)0,
                        cAcc[msub][nsub], false, false);
                }
            }
            waitAsync();      // own wave's async copies landed
            __syncthreads();  // visible to all waves; old buf free
        }

        // fold expert into running output: out += su * (dot + bias)
#pragma unroll
        for (int msub = 0; msub < 4; ++msub) {
#pragma unroll
            for (int nsub = 0; nsub < 2; ++nsub) {
                int oL = nw + nsub * 16 + laneM;
                float b = bexpLds[oL];
#pragma unroll
                for (int i = 0; i < 8; ++i) {
                    int rL = mw + msub * 16 + mHalf + i;
                    float sc = suLds[rL * E + e];
                    outAcc[msub][nsub][i] += sc * (cAcc[msub][nsub][i] + b);
                }
            }
        }
    }

#pragma unroll
    for (int msub = 0; msub < 4; ++msub) {
#pragma unroll
        for (int nsub = 0; nsub < 2; ++nsub) {
            int oL = nw + nsub * 16 + laneM;
#pragma unroll
            for (int i = 0; i < 8; ++i) {
                int rL = mw + msub * 16 + mHalf + i;
                out[(size_t)(mBase + rL) * O + oBase + oL] = outAcc[msub][nsub][i];
            }
        }
    }
}

// =====================================================================
// launcher
// =====================================================================
extern "C" void kernel_launch(void* const* d_in, const int* in_sizes, int n_in,
                              void* d_out, int out_size, void* d_ws, size_t ws_size,
                              hipStream_t stream) {
    const float* x    = (const float*)d_in[0];
    const float* Wsel = (const float*)d_in[1];
    const float* bsel = (const float*)d_in[2];
    const float* Wexp = (const float*)d_in[3];
    const float* bexp = (const float*)d_in[4];

    float* out      = (float*)d_out;            // [N*O]
    float* softCost = out + (size_t)N * O;      // [N]

    char* ws = (char*)d_ws;
    float*    w      = (float*)(ws);                           // N*E
    float*    sw2    = (float*)(ws + 262144);                  // N*E
    float*    su     = (float*)(ws + 524288);                  // N*E
    float*    colsum = (float*)(ws + 786432);                  // E
    unsigned* u1bits = (unsigned*)(ws + 786432 + 256);         // N
    unsigned* u2bits = (unsigned*)(ws + 786432 + 256 + 16384); // N
    bf16_t*   xb     = (bf16_t*)(ws + (1u << 20));             // N*D bf16 (8 MB)
    bf16_t*   Wb     = (bf16_t*)(ws + (9u << 20));             // E*O*D bf16 (32 MB)

    (void)hipMemsetAsync(u2bits, 0, (size_t)N * sizeof(unsigned), stream);

    selector_kernel<<<N / 8, 256, 0, stream>>>(x, Wsel, bsel, w, sw2, u1bits, softCost);
    colsum_kernel<<<E, 256, 0, stream>>>(w, colsum);
    expert_sort_kernel<<<E, 1024, 0, stream>>>(w, colsum, u2bits);
    finalize_su<<<(N * E) / 256, 256, 0, stream>>>(sw2, u1bits, u2bits, su);

    cvt_bf16_kernel<<<(N * D / 8) / 256, 256, 0, stream>>>(x, xb, N * D / 8);
    cvt_bf16_kernel<<<(E * O * D / 8) / 256, 256, 0, stream>>>(Wexp, Wb, E * O * D / 8);

    dim3 grid(O / BN, N / BM);  // (8, 32)
    moe_gemm<<<grid, 256, 0, stream>>>(xb, Wb, bexp, su, out);
}
